// MultiheadAttention_89962384982639
// MI455X (gfx1250) — matrix operand order repaired and checked by reading, not compiled, not run on hardware
//
#include <hip/hip_runtime.h>
#include <hip/hip_bf16.h>

// ---------------------------------------------------------------------------
// CDNA5 (gfx1250) fused multi-head attention with batch-axis softmax quirk.
// Matmuls via v_wmma_f32_16x16x32_f16 (wave32); A-operand staging through
// GLOBAL_LOAD_ASYNC_TO_LDS_B128 (ASYNCcnt) in the dense GEMM kernels.
// ---------------------------------------------------------------------------

typedef __attribute__((ext_vector_type(16))) _Float16 v16h;
typedef __attribute__((ext_vector_type(8)))  float    v8f;
typedef __attribute__((ext_vector_type(4)))  _Float16 v4h;

// D = A(16x32 f16) * B(32x16 f16) + C(16x16 f32)
__device__ __forceinline__ v8f wmma_f16(v16h a, v16h b, v8f c) {
    return __builtin_amdgcn_wmma_f32_16x16x32_f16(
        /*neg_a=*/false, a, /*neg_b=*/false, b,
        /*c_mod=*/(short)0, c, /*reuse_a=*/false, /*reuse_b=*/false);
}

// ---- Fragment loaders (CDNA5 ISA §7.12.2 layouts, wave32) -------------
// A 16x32 f16, row-major source (ld halves). Lane l: m=l&15, g=l>>4.
// Per lane: halves [8g..8g+7] and [16+8g..16+8g+7] of row m.
__device__ __forceinline__ v16h load_a16(const _Float16* src, int ld, int lane) {
    const int m = lane & 15, g = lane >> 4;
    const _Float16* row = src + (size_t)m * ld + 8 * g;
    v16h out;
    *(float4*)&out       = *(const float4*)(row);        // K 8g..8g+7
    *((float4*)&out + 1) = *(const float4*)(row + 16);   // K 16+8g..+7
    return out;
}

// B 32x16 f16 from a TRANSPOSED source srcT[N][K] row-major (ld halves).
// B layout matching A: lane holds K [8g..8g+7] and [16+8g..16+8g+7]; n=l&15.
__device__ __forceinline__ v16h load_bT(const _Float16* srcT, int ld, int lane) {
    const int n = lane & 15, g = lane >> 4;
    const _Float16* p = srcT + (size_t)n * ld + 8 * g;
    v16h out;
    *(float4*)&out       = *(const float4*)(p);
    *((float4*)&out + 1) = *(const float4*)(p + 16);
    return out;
}

// ---- Async global->LDS copy of one 16B chunk per lane (GV mode) ------------
// lds_off: byte offset inside group segment (= low 32 bits of generic addr).
__device__ __forceinline__ void async_copy_b128(const _Float16* gaddr, void* lds_ptr) {
    unsigned lds_off = (unsigned)(size_t)lds_ptr;          // LDS aperture low bits
    unsigned long long ga = (unsigned long long)(size_t)gaddr;
    asm volatile("global_load_async_to_lds_b128 %0, %1, off"
                 :: "v"(lds_off), "v"(ga) : "memory");
}
__device__ __forceinline__ void async_wait0() {
    asm volatile("s_wait_asynccnt 0x0" ::: "memory");
}

// ---------------------------------------------------------------------------
// Kernel 0: f32 -> f16 convert of activations (removes cvts from GEMM loops).
// ---------------------------------------------------------------------------
__global__ void k_cvt_f16(const float* __restrict__ in, _Float16* __restrict__ out,
                          int ntotal) {
    const int idx = (blockIdx.x * 256 + threadIdx.x) * 4;
    if (idx >= ntotal) return;
    float4 v = *(const float4*)(in + idx);
    v4h o;
    o[0] = (_Float16)v.x; o[1] = (_Float16)v.y;
    o[2] = (_Float16)v.z; o[3] = (_Float16)v.w;
    *(v4h*)(out + idx) = o;
}

// ---------------------------------------------------------------------------
// Kernel 1: transpose + f32->f16 convert.  in[h][d][s] -> out[h][s][d]
// (also used for o_w with h=1, d=1024, s=1024)
// ---------------------------------------------------------------------------
__global__ void k_transpose_cvt(const float* __restrict__ in,
                                _Float16* __restrict__ out, int D, int S) {
    const int h = blockIdx.y;
    const int idx = blockIdx.x * 256 + threadIdx.x;
    const int total = D * S;
    if (idx >= total) return;
    const int d = idx / S, s = idx % S;
    out[(size_t)h * total + (size_t)s * D + d] = (_Float16)in[(size_t)h * total + idx];
}

// ---------------------------------------------------------------------------
// Kernel 2: projection GEMM.  Out[b,h,i,s] = sum_d X16[b,i,d] * W[h,d,s]
// X16 is pre-converted f16; A slab (32 rows x 256 K) async-staged into LDS.
// grid (32 i-tiles, 16 h, 8 b), 256 thr = 8 waves (2 m-subtiles x 4 n-subtiles).
// transpose_out=1 stores Out as [b][h][s][i] (for V).
// ---------------------------------------------------------------------------
__global__ void k_proj(const _Float16* __restrict__ X16, const _Float16* __restrict__ WT,
                       _Float16* __restrict__ Out, int transpose_out) {
    const int I = 1024, D = 1024, S = 64;
    const int KB = 256;                                    // K slab
    __shared__ __align__(16) _Float16 Xsl[32 * KB];        // 16 KB
    const int it = blockIdx.x, h = blockIdx.y, b = blockIdx.z;
    const int tid = threadIdx.x;
    const int lane = tid & 31, w = tid >> 5;
    const int mi = w & 1, nt = w >> 1;
    const int m0 = it * 32 + mi * 16, n0 = nt * 16;
    const _Float16* Xrows = X16 + (size_t)b * I * D + (size_t)(it * 32) * D;
    const _Float16* WTh = WT + (size_t)h * D * S;          // [S][D], ld=D
    v8f acc = {0.f,0.f,0.f,0.f,0.f,0.f,0.f,0.f};
    for (int kb = 0; kb < D; kb += KB) {
        // stage 32 x 256 f16 slab: 4 x b128 per thread, async to LDS
#pragma unroll
        for (int ch = 0; ch < 4; ++ch) {
            const int f = tid * 8 + ch * 2048;             // half index in slab
            const int r = f / KB, c = f % KB;
            async_copy_b128(Xrows + (size_t)r * D + kb + c, &Xsl[f]);
        }
        async_wait0();
        __syncthreads();
#pragma unroll
        for (int kc = 0; kc < KB / 32; ++kc) {
            v16h a  = load_a16(&Xsl[(mi * 16) * KB] + 32 * kc, KB, lane);
            v16h bt = load_bT(WTh + (size_t)n0 * D + kb + 32 * kc, D, lane);
            acc = wmma_f16(a, bt, acc);
        }
        __syncthreads();
    }
    const int n = lane & 15, g = lane >> 4;
    if (!transpose_out) {
        _Float16* Obh = Out + ((size_t)b * 16 + h) * I * S;   // [i][s] ld=S
#pragma unroll
        for (int r = 0; r < 8; ++r)
            Obh[(size_t)(m0 + r + 8 * g) * S + n0 + n] = (_Float16)acc[r];
    } else {
        _Float16* Obh = Out + ((size_t)b * 16 + h) * S * I;   // [s][i] ld=I
#pragma unroll
        for (int r = 0; r < 8; ++r)
            Obh[(size_t)(n0 + n) * I + (m0 + r + 8 * g)] = (_Float16)acc[r];
    }
}

// ---------------------------------------------------------------------------
// Kernel 3: fused attention with softmax over the BATCH axis.
// grid (64 i-tiles, 16 h), 256 thr = 8 waves; wave w owns batch b=w.
// ---------------------------------------------------------------------------
__global__ void k_attn(const _Float16* __restrict__ WQ, const _Float16* __restrict__ WK,
                       const _Float16* __restrict__ WVT, _Float16* __restrict__ Heads) {
    const int I = 1024, J = 1024, S = 64, H = 16;
    __shared__ float    lds_x[8][512];                 // scaled scores [b][m*32+j]
    __shared__ __align__(16) _Float16 lds_w[8][512];   // softmax weights
    const int it = blockIdx.x, h = blockIdx.y;
    const int lane = threadIdx.x & 31, w = threadIdx.x >> 5;   // w = batch b
    const int i0 = it * 16;
    const int n = lane & 15, g = lane >> 4;
    const size_t bh = (size_t)w * H + h;
    const _Float16* Q  = WQ  + bh * I * S + (size_t)i0 * S;   // [16][64]
    const _Float16* Kp = WK  + bh * J * S;                    // [j][s] ld=S
    const _Float16* VT = WVT + bh * S * J;                    // [s][j] ld=J

    v16h qa0 = load_a16(Q + 0,  S, lane);
    v16h qa1 = load_a16(Q + 32, S, lane);

    v8f zero = {0.f,0.f,0.f,0.f,0.f,0.f,0.f,0.f};
    v8f hacc[4] = {zero, zero, zero, zero};            // 16 i x 64 t accumulators

    for (int j0 = 0; j0 < J; j0 += 32) {
        // ---- scores: two 16x16 tiles, K=64 over two chunks -----------------
#pragma unroll
        for (int jt = 0; jt < 2; ++jt) {
            v16h kb0 = load_bT(Kp + (size_t)(j0 + jt * 16) * S + 0,  S, lane);
            v16h kb1 = load_bT(Kp + (size_t)(j0 + jt * 16) * S + 32, S, lane);
            v8f sc = zero;
            sc = wmma_f16(qa0, kb0, sc);
            sc = wmma_f16(qa1, kb1, sc);
#pragma unroll
            for (int r = 0; r < 8; ++r)                 // scale = 1/sqrt(H) quirk
                lds_x[w][(r + 8 * g) * 32 + jt * 16 + n] = 0.25f * sc[r];
        }
        __syncthreads();
        // ---- softmax across b (8 values per element), elementwise ----------
#pragma unroll
        for (int rep = 0; rep < 2; ++rep) {
            const int idx = threadIdx.x + rep * 256;
            float xv[8], m = -3.4e38f;
#pragma unroll
            for (int bb = 0; bb < 8; ++bb) { xv[bb] = lds_x[bb][idx]; m = fmaxf(m, xv[bb]); }
            float e[8], ssum = 0.f;
#pragma unroll
            for (int bb = 0; bb < 8; ++bb) { e[bb] = __expf(xv[bb] - m); ssum += e[bb]; }
            const float inv = 1.0f / ssum;
#pragma unroll
            for (int bb = 0; bb < 8; ++bb) lds_w[bb][idx] = (_Float16)(e[bb] * inv);
        }
        __syncthreads();
        // ---- heads += weight(16x32) * V(32x64) -----------------------------
        v16h wa = load_a16(&lds_w[w][0], 32, lane);
#pragma unroll
        for (int tt = 0; tt < 4; ++tt) {
            v16h vb = load_bT(VT + (size_t)(tt * 16) * J + j0, J, lane);
            hacc[tt] = wmma_f16(wa, vb, hacc[tt]);
        }
        __syncthreads();   // protect LDS before next iteration overwrites
    }
    _Float16* Hb = Heads + bh * I * S;                 // [i][t] ld=S
#pragma unroll
    for (int tt = 0; tt < 4; ++tt)
#pragma unroll
        for (int r = 0; r < 8; ++r)
            Hb[(size_t)(i0 + r + 8 * g) * S + tt * 16 + n] = (_Float16)hacc[tt][r];
}

// ---------------------------------------------------------------------------
// Kernel 4: output GEMM. The torch .view quirk means flat[b] is literally
// heads[b] reinterpreted as row-major [1024][1024]. Out = flat @ o_w.
// A slab (16 rows x 256 K, shared by all 8 waves) async-staged into LDS.
// grid (8 o-tiles, 64 i-tiles, 8 b), 256 thr = 8 waves (each 16 cols).
// ---------------------------------------------------------------------------
__global__ void k_out(const _Float16* __restrict__ Flat, const _Float16* __restrict__ OWT,
                      float* __restrict__ Out) {
    const int M = 1024, N = 1024, Kd = 1024;
    const int KB = 256;
    __shared__ __align__(16) _Float16 Asl[16 * KB];    // 8 KB
    const int ot = blockIdx.x, it = blockIdx.y, b = blockIdx.z;
    const int tid = threadIdx.x;
    const int lane = tid & 31, w = tid >> 5;
    const int m0 = it * 16, n0 = ot * 128 + w * 16;
    const _Float16* Arows = Flat + (size_t)b * M * Kd + (size_t)m0 * Kd;
    v8f acc = {0.f,0.f,0.f,0.f,0.f,0.f,0.f,0.f};
    for (int kb = 0; kb < Kd; kb += KB) {
        // stage 16 x 256 f16 slab: 2 x b128 per thread, async to LDS
#pragma unroll
        for (int ch = 0; ch < 2; ++ch) {
            const int f = tid * 8 + ch * 2048;          // half index in slab
            const int r = f / KB, c = f % KB;
            async_copy_b128(Arows + (size_t)r * Kd + kb + c, &Asl[f]);
        }
        async_wait0();
        __syncthreads();
#pragma unroll
        for (int kc = 0; kc < KB / 32; ++kc) {
            v16h a  = load_a16(&Asl[0] + 32 * kc, KB, lane);           // ds_load
            v16h bt = load_bT(OWT + (size_t)n0 * Kd + kb + 32 * kc, Kd, lane);
            acc = wmma_f16(a, bt, acc);
        }
        __syncthreads();
    }
    float* Ob = Out + (size_t)b * M * N;
    const int n = lane & 15, g = lane >> 4;
#pragma unroll
    for (int r = 0; r < 8; ++r)
        Ob[(size_t)(m0 + r + 8 * g) * N + n0 + n] = acc[r];
}

// ---------------------------------------------------------------------------
extern "C" void kernel_launch(void* const* d_in, const int* in_sizes, int n_in,
                              void* d_out, int out_size, void* d_ws, size_t ws_size,
                              hipStream_t stream) {
    const float* query = (const float*)d_in[0];   // [8,1024,1024]
    const float* key   = (const float*)d_in[1];   // [8,1024,1024]
    const float* value = (const float*)d_in[2];   // [8,1024,1024]
    const float* q_w   = (const float*)d_in[3];   // [16,1024,64]
    const float* k_w   = (const float*)d_in[4];   // [16,1024,64]
    const float* v_w   = (const float*)d_in[5];   // [16,1024,64]
    const float* o_w   = (const float*)d_in[6];   // [1024,1024]
    float* out = (float*)d_out;                   // [8,1024,1024]

    // Workspace layout (f16), ~120 MB total
    char* ws = (char*)d_ws;
    const size_t PROJ_BYTES = (size_t)8 * 16 * 1024 * 64 * 2;   // 16 MB
    const size_t W_BYTES    = (size_t)16 * 1024 * 64 * 2;       //  2 MB
    const size_t OW_BYTES   = (size_t)1024 * 1024 * 2;          //  2 MB
    const size_t X_BYTES    = (size_t)8 * 1024 * 1024 * 2;      // 16 MB
    _Float16* wq    = (_Float16*)(ws + 0 * PROJ_BYTES);
    _Float16* wk    = (_Float16*)(ws + 1 * PROJ_BYTES);
    _Float16* wvT   = (_Float16*)(ws + 2 * PROJ_BYTES);   // [b][h][s][j]
    _Float16* heads = (_Float16*)(ws + 3 * PROJ_BYTES);
    char* wpart = ws + 4 * PROJ_BYTES;
    _Float16* qwT = (_Float16*)(wpart);                   // [h][s][d]
    _Float16* kwT = (_Float16*)(wpart + 1 * W_BYTES);
    _Float16* vwT = (_Float16*)(wpart + 2 * W_BYTES);
    _Float16* owT = (_Float16*)(wpart + 3 * W_BYTES);     // [o][t]
    char* xpart = wpart + 3 * W_BYTES + OW_BYTES;
    _Float16* q16 = (_Float16*)(xpart + 0 * X_BYTES);
    _Float16* k16 = (_Float16*)(xpart + 1 * X_BYTES);
    _Float16* v16 = (_Float16*)(xpart + 2 * X_BYTES);
    (void)in_sizes; (void)n_in; (void)out_size; (void)ws_size;

    const int XN = 8 * 1024 * 1024;
    // 0) convert activations to f16 once
    k_cvt_f16<<<dim3(XN / 1024), 256, 0, stream>>>(query, q16, XN);
    k_cvt_f16<<<dim3(XN / 1024), 256, 0, stream>>>(key,   k16, XN);
    k_cvt_f16<<<dim3(XN / 1024), 256, 0, stream>>>(value, v16, XN);

    // 1) weight transposes + f32->f16
    k_transpose_cvt<<<dim3(256, 16), 256, 0, stream>>>(q_w, qwT, 1024, 64);
    k_transpose_cvt<<<dim3(256, 16), 256, 0, stream>>>(k_w, kwT, 1024, 64);
    k_transpose_cvt<<<dim3(256, 16), 256, 0, stream>>>(v_w, vwT, 1024, 64);
    k_transpose_cvt<<<dim3(4096, 1), 256, 0, stream>>>(o_w, owT, 1024, 1024);

    // 2) projections (V stored transposed for B-operand loads)
    k_proj<<<dim3(32, 16, 8), 256, 0, stream>>>(q16, qwT, wq, 0);
    k_proj<<<dim3(32, 16, 8), 256, 0, stream>>>(k16, kwT, wk, 0);
    k_proj<<<dim3(32, 16, 8), 256, 0, stream>>>(v16, vwT, wvT, 1);

    // 3) fused scores -> batch-softmax -> heads
    k_attn<<<dim3(64, 16), 256, 0, stream>>>(wq, wk, wvT, heads);

    // 4) output GEMM (heads buffer IS the .view-reshaped flat matrix)
    k_out<<<dim3(8, 64, 8), 256, 0, stream>>>(heads, owT, out);
}
